// ChannelWise_74337293959379
// MI455X (gfx1250) — matrix operand order repaired and checked by reading, not compiled, not run on hardware
//
#include <hip/hip_runtime.h>
#include <hip/hip_bf16.h>
#include <math.h>

typedef __bf16 bf16_t;
typedef __attribute__((ext_vector_type(16))) __bf16 v16bf;
typedef __attribute__((ext_vector_type(8)))  __bf16 v8bf;
typedef __attribute__((ext_vector_type(8)))  float   v8f;
typedef __attribute__((ext_vector_type(4)))  int     v4i_t;

#define BATCH 4
#define CCH   256
#define HGT   64
#define WID   64
#define NPIX  4096      /* H*W */
#define KDIM  512       /* 2C  */
#define MOUT  320       /* 32 + 32 + 256 output channels of fused projection */
#define C8    32

#if defined(__has_builtin)
#if __has_builtin(__builtin_amdgcn_global_load_async_to_lds_b128)
#define HAVE_ASYNC_LDS 1
#endif
#endif

/* ---------------- workspace layout (bytes) ---------------- */
constexpr size_t alignup(size_t x) { return (x + 255) & ~size_t(255); }
constexpr size_t OFF_CAT  = 0;
constexpr size_t SZ_CAT   = (size_t)BATCH * NPIX * KDIM * 2;          /* 16 MB bf16 */
constexpr size_t OFF_W    = alignup(OFF_CAT + SZ_CAT);
constexpr size_t SZ_W     = (size_t)MOUT * KDIM * 2;
constexpr size_t OFF_BIAS = alignup(OFF_W + SZ_W);
constexpr size_t SZ_BIAS  = (size_t)MOUT * 4;
constexpr size_t OFF_FB   = alignup(OFF_BIAS + SZ_BIAS);
constexpr size_t SZ_FB    = (size_t)BATCH * NPIX * C8 * 2;            /* 1 MB */
constexpr size_t OFF_FC   = alignup(OFF_FB + SZ_FB);
constexpr size_t OFF_FD   = alignup(OFF_FC + SZ_FB);                  /* feat_d^T 8 MB */

static __device__ __forceinline__ v16bf join16(v8bf lo, v8bf hi) {
  v16bf r;
#pragma unroll
  for (int i = 0; i < 8; ++i) { r[i] = lo[i]; r[i + 8] = hi[i]; }
  return r;
}
static __device__ __forceinline__ v8f vzero8() {
  v8f z;
#pragma unroll
  for (int i = 0; i < 8; ++i) z[i] = 0.f;
  return z;
}

/* ============ kernel 1: fuse + convert weights to bf16 ============ */
__global__ void k_cvt_weights(const float* __restrict__ wb, const float* __restrict__ wc,
                              const float* __restrict__ wd, const float* __restrict__ bb,
                              const float* __restrict__ bc, const float* __restrict__ bd,
                              bf16_t* __restrict__ W, float* __restrict__ bias) {
  int t = blockIdx.x * blockDim.x + threadIdx.x;
  if (t < MOUT * KDIM) {
    int row = t / KDIM, col = t % KDIM;
    float v = (row < 32) ? wb[row * KDIM + col]
            : (row < 64) ? wc[(row - 32) * KDIM + col]
                         : wd[(row - 64) * KDIM + col];
    W[t] = (bf16_t)v;
  }
  if (t < MOUT) {
    bias[t] = (t < 32) ? bb[t] : (t < 64) ? bc[t - 32] : bd[t - 64];
  }
}

/* ============ kernel 2: 3x3 avg/max pool -> cat[B][N][512] bf16 ============ */
__global__ void k_pool_cat(const float* __restrict__ x, bf16_t* __restrict__ cat) {
  int t = blockIdx.x * blockDim.x + threadIdx.x;  /* c fastest for coalesced bf16 writes */
  int c = t & (CCH - 1);
  int n = (t >> 8) & (NPIX - 1);
  int b = t >> 20;
  int h = n >> 6, w = n & 63;
  const float* xp = x + ((size_t)b * CCH + c) * (HGT * WID);
  float s = 0.f, m = -__builtin_inff();
#pragma unroll
  for (int dh = -1; dh <= 1; ++dh) {
#pragma unroll
    for (int dw = -1; dw <= 1; ++dw) {
      int hh = h + dh, ww = w + dw;
      if (hh >= 0 && hh < HGT && ww >= 0 && ww < WID) {
        float v = xp[hh * WID + ww];
        s += v;
        m = fmaxf(m, v);
      }
    }
  }
  bf16_t* cp = cat + ((size_t)b * NPIX + n) * KDIM;
  cp[c]       = (bf16_t)(s * (1.f / 9.f)); /* count_include_pad: always /9 */
  cp[CCH + c] = (bf16_t)m;                 /* -inf pad: max over valid     */
}

/* ============ kernel 3: feat = W @ cat + bias (WMMA bf16) ============ */
/* one wave -> one 16(out-ch) x 16(pixel) tile, K = 512 unrolled in 16 WMMAs */
__global__ void __launch_bounds__(256)
k_proj(const bf16_t* __restrict__ W, const float* __restrict__ bias,
       const bf16_t* __restrict__ cat, bf16_t* __restrict__ fb,
       bf16_t* __restrict__ fc, bf16_t* __restrict__ fdt) {
  const int lane = threadIdx.x & 31;
  const int gw   = blockIdx.x * (blockDim.x >> 5) + (threadIdx.x >> 5);
  const int mt   = gw % 20;           /* 20 M-tiles of 16 = 320 out channels */
  const int pt   = (gw / 20) & 255;   /* 256 pixel tiles of 16               */
  const int b    = gw / (20 * 256);
  const int lc   = lane & 15;
  const int sel  = lane >> 4;

  const bf16_t* arow = W + (size_t)(mt * 16 + lc) * KDIM;                  /* A: row = lane%16 */
  const bf16_t* brow = cat + ((size_t)b * NPIX + pt * 16 + lc) * KDIM;     /* B: col = pixel   */
  v8f c = vzero8();
#pragma unroll
  for (int k = 0; k < KDIM; k += 32) {
    v8bf a0 = *(const v8bf*)(arow + k + sel * 16);
    v8bf a1 = *(const v8bf*)(arow + k + sel * 16 + 8);
    v8bf b0 = *(const v8bf*)(brow + k + sel * 16);
    v8bf b1 = *(const v8bf*)(brow + k + sel * 16 + 8);
    c = __builtin_amdgcn_wmma_f32_16x16x32_bf16(false, join16(a0, a1),
                                                false, join16(b0, b1),
                                                (short)0, c, false, false);
  }
  const float* bp  = bias + mt * 16 + sel * 8;   /* C rows = r + 8*sel */
  const int    pix = pt * 16 + lc;
  if (mt < 4) {  /* feat_b (mt 0,1) / feat_c (mt 2,3): layout [B][N][32] */
    v8bf o;
#pragma unroll
    for (int r = 0; r < 8; ++r) o[r] = (bf16_t)(c[r] + bp[r]);
    bf16_t* dst = (mt < 2 ? fb : fc) +
                  ((size_t)b * NPIX + pix) * C8 + (mt & 1) * 16 + sel * 8;
    *(v8bf*)dst = o;
  } else {       /* feat_d transposed: [B][256][N] */
#pragma unroll
    for (int r = 0; r < 8; ++r) {
      int ch = (mt - 4) * 16 + sel * 8 + r;
      fdt[((size_t)b * CCH + ch) * NPIX + pix] = (bf16_t)(c[r] + bp[r]);
    }
  }
}

/* ============ kernel 4: flash attention + alpha blend ============ */
/* block = 256 thr (8 waves); wave owns 16 query rows x 128 v-channels.
   V tiles staged once per block via async global->LDS (double buffered),
   overlapped with softmax of the current tile.                         */
#define PSTR 72   /* padded LDS row stride (bf16 elems), keeps 16B align */

#if HAVE_ASYNC_LDS
/* copy one V tile: 256 vch x 64 keys bf16 (32KB) as 2048 x 16B chunks */
static __device__ __forceinline__ void async_copy_vtile(const bf16_t* __restrict__ fdb,
                                                        bf16_t* lbuf, int ktn, int tid) {
#pragma unroll
  for (int ci = tid; ci < CCH * 8; ci += 256) {   /* 8 ops per lane */
    const int vch = ci >> 3, kg = ci & 7;
    const bf16_t* g = fdb + (size_t)vch * NPIX + ktn + kg * 8;
    bf16_t* l = lbuf + vch * 64 + kg * 8;
    __builtin_amdgcn_global_load_async_to_lds_b128(
        (__attribute__((address_space(1))) v4i_t*)g,
        (__attribute__((address_space(3))) v4i_t*)l, 0, 0);
  }
}
#endif

__global__ void __launch_bounds__(256, 1)
k_attn(const bf16_t* __restrict__ fb, const bf16_t* __restrict__ fc,
       const bf16_t* __restrict__ fdt, const float* __restrict__ x,
       const float* __restrict__ alphap, float* __restrict__ out) {
#if HAVE_ASYNC_LDS
  __shared__ bf16_t lds_v[2][CCH * 64];     /* 2 x 32KB V tiles [vch][key] */
#endif
  __shared__ bf16_t lds_p[8 * 16 * PSTR];   /* wave-private P tiles */
  const int lane = threadIdx.x & 31;
  const int wave = threadIdx.x >> 5;
  const int wq   = wave & 3;   /* query sub-tile  */
  const int wv   = wave >> 2;  /* v-channel half  */
  const int b    = blockIdx.x >> 6;
  const int qt   = blockIdx.x & 63;
  const int lc   = lane & 15;
  const int sel  = lane >> 4;
  const int qbase = qt * 64 + wq * 16;

  /* Q as WMMA A operand: row = lane%16 = query, 32 dk values */
  const bf16_t* qrow = fb + ((size_t)b * NPIX + qbase + lc) * C8;
  v16bf aq = join16(*(const v8bf*)(qrow + sel * 16),
                    *(const v8bf*)(qrow + sel * 16 + 8));

  v8f acc[8];
#pragma unroll
  for (int t = 0; t < 8; ++t) acc[t] = vzero8();
  float mrun[8], lrun[8];
#pragma unroll
  for (int r = 0; r < 8; ++r) { mrun[r] = -__builtin_inff(); lrun[r] = 0.f; }

  bf16_t* pbuf = lds_p + wave * 16 * PSTR;
  const bf16_t* fcb = fc + (size_t)b * NPIX * C8;
  const bf16_t* fdb = fdt + (size_t)b * CCH * NPIX;

#if HAVE_ASYNC_LDS
  async_copy_vtile(fdb, &lds_v[0][0], 0, threadIdx.x);   /* prime buffer 0 */
#endif

  for (int kt = 0; kt < NPIX; kt += 64) {
#if HAVE_ASYNC_LDS
    const int  cur     = (kt >> 6) & 1;
    const bool hasnext = (kt + 64) < NPIX;
    if (hasnext)
      async_copy_vtile(fdb, &lds_v[cur ^ 1][0], kt + 64, threadIdx.x);
#endif
    /* --- scores S = Q.K^T : 4 tiles of 16 keys, K=32 in one WMMA each --- */
    v8f s[4];
#pragma unroll
    for (int kk = 0; kk < 4; ++kk) {
      const bf16_t* krow = fcb + (size_t)(kt + kk * 16 + lc) * C8 + sel * 16;
      v16bf bk = join16(*(const v8bf*)krow, *(const v8bf*)(krow + 8));
      s[kk] = __builtin_amdgcn_wmma_f32_16x16x32_bf16(false, aq, false, bk,
                                                      (short)0, vzero8(), false, false);
    }
    /* --- online softmax row statistics (rows live in 16-lane groups) --- */
    float tmax[8];
#pragma unroll
    for (int r = 0; r < 8; ++r)
      tmax[r] = fmaxf(fmaxf(s[0][r], s[1][r]), fmaxf(s[2][r], s[3][r]));
#pragma unroll
    for (int r = 0; r < 8; ++r) {
      tmax[r] = fmaxf(tmax[r], __shfl_xor(tmax[r], 1, 32));
      tmax[r] = fmaxf(tmax[r], __shfl_xor(tmax[r], 2, 32));
      tmax[r] = fmaxf(tmax[r], __shfl_xor(tmax[r], 4, 32));
      tmax[r] = fmaxf(tmax[r], __shfl_xor(tmax[r], 8, 32));
    }
    float scl[8], rsum[8];
#pragma unroll
    for (int r = 0; r < 8; ++r) {
      float nm = fmaxf(mrun[r], tmax[r]);
      scl[r]  = __expf(mrun[r] - nm);
      mrun[r] = nm;
      rsum[r] = 0.f;
    }
    /* P = exp(S - m), spill to LDS in row-major [16 q][64 key] bf16 */
#pragma unroll
    for (int kk = 0; kk < 4; ++kk)
#pragma unroll
      for (int r = 0; r < 8; ++r) {
        float p = __expf(s[kk][r] - mrun[r]);
        rsum[r] += p;
        pbuf[(r + sel * 8) * PSTR + kk * 16 + lc] = (bf16_t)p;
      }
#pragma unroll
    for (int r = 0; r < 8; ++r) {
      rsum[r] += __shfl_xor(rsum[r], 1, 32);
      rsum[r] += __shfl_xor(rsum[r], 2, 32);
      rsum[r] += __shfl_xor(rsum[r], 4, 32);
      rsum[r] += __shfl_xor(rsum[r], 8, 32);
      lrun[r] = lrun[r] * scl[r] + rsum[r];
    }
#pragma unroll
    for (int t = 0; t < 8; ++t)
#pragma unroll
      for (int r = 0; r < 8; ++r) acc[t][r] = acc[t][r] * scl[r];

    asm volatile("s_wait_dscnt 0" ::: "memory"); /* LDS P stores -> loads */

    /* re-load P in WMMA A layout (two K=32 chunks) */
    v16bf ap[2];
#pragma unroll
    for (int ks = 0; ks < 2; ++ks) {
      const bf16_t* pr = pbuf + lc * PSTR + ks * 32 + sel * 16;
      ap[ks] = join16(*(const v8bf*)pr, *(const v8bf*)(pr + 8));
    }
    /* prefetch next K tile into near cache */
    if (kt + 64 < NPIX)
      __builtin_prefetch((const void*)(fcb + (size_t)(kt + 64 + lc) * C8), 0, 0);

#if HAVE_ASYNC_LDS
    /* current V tile resident; next tile may still be in flight (<=8 ops) */
    if (hasnext) asm volatile("s_wait_asynccnt 8" ::: "memory");
    else         asm volatile("s_wait_asynccnt 0" ::: "memory");
    __syncthreads();
#endif

    /* --- O += P.V : 8 v-channel tiles x 2 key chunks = 16 WMMAs --- */
#pragma unroll
    for (int t = 0; t < 8; ++t) {
      const int vch = wv * 128 + t * 16 + lc;
#if HAVE_ASYNC_LDS
      const bf16_t* vb = &lds_v[cur][0] + vch * 64 + sel * 16;
#else
      const bf16_t* vb = fdb + (size_t)vch * NPIX + kt + sel * 16;
#endif
#pragma unroll
      for (int ks = 0; ks < 2; ++ks) {
        v16bf bv = join16(*(const v8bf*)(vb + ks * 32),
                          *(const v8bf*)(vb + ks * 32 + 8));
        acc[t] = __builtin_amdgcn_wmma_f32_16x16x32_bf16(false, ap[ks], false, bv,
                                                         (short)0, acc[t], false, false);
      }
    }
#if HAVE_ASYNC_LDS
    __syncthreads();  /* protect buffer cur before iteration kt+64 refills it */
#endif
  }

  /* --- epilogue: normalize, blend with x, store fp32 --- */
  const float alpha = *alphap;
  float inv[8];
#pragma unroll
  for (int r = 0; r < 8; ++r) inv[r] = 1.f / lrun[r];
#pragma unroll
  for (int t = 0; t < 8; ++t) {
    const int vch = wv * 128 + t * 16 + lc;
#pragma unroll
    for (int r = 0; r < 8; ++r) {
      const int q = qbase + sel * 8 + r;
      const size_t idx = ((size_t)b * CCH + vch) * NPIX + q;
      out[idx] = alpha * (acc[t][r] * inv[r]) + (1.f - alpha) * x[idx];
    }
  }
}

/* ==================== host launcher ==================== */
extern "C" void kernel_launch(void* const* d_in, const int* in_sizes, int n_in,
                              void* d_out, int out_size, void* d_ws, size_t ws_size,
                              hipStream_t stream) {
  (void)in_sizes; (void)n_in; (void)out_size; (void)ws_size;
  const float* x  = (const float*)d_in[0];
  const float* wb = (const float*)d_in[1];
  const float* bb = (const float*)d_in[2];
  const float* wc = (const float*)d_in[3];
  const float* bc = (const float*)d_in[4];
  const float* wd = (const float*)d_in[5];
  const float* bd = (const float*)d_in[6];
  const float* alpha = (const float*)d_in[7];
  float* out = (float*)d_out;

  char* ws = (char*)d_ws;
  bf16_t* cat  = (bf16_t*)(ws + OFF_CAT);
  bf16_t* W    = (bf16_t*)(ws + OFF_W);
  float*  bias = (float*) (ws + OFF_BIAS);
  bf16_t* fb   = (bf16_t*)(ws + OFF_FB);
  bf16_t* fc   = (bf16_t*)(ws + OFF_FC);
  bf16_t* fdt  = (bf16_t*)(ws + OFF_FD);

  k_cvt_weights<<<(MOUT * KDIM + 255) / 256, 256, 0, stream>>>(wb, wc, wd, bb, bc, bd, W, bias);
  k_pool_cat<<<(BATCH * NPIX * CCH) / 256, 256, 0, stream>>>(x, cat);
  k_proj<<<(BATCH * 20 * 256) / 8, 256, 0, stream>>>(W, bias, cat, fb, fc, fdt);
  k_attn<<<BATCH * 64, 256, 0, stream>>>(fb, fc, fdt, x, alpha, out);
}
